// GatedDeltaNet_29643864277429
// MI455X (gfx1250) — compile-verified, matrix operand order checked by reading, not run on hardware
//
#include <hip/hip_runtime.h>

// ---------------- types ----------------
typedef __attribute__((ext_vector_type(16))) __bf16 v16bf;
typedef __attribute__((ext_vector_type(8)))  float  v8f;

#define B_    2
#define S_    4096
#define D_    2048
#define H_    16
#define KVH_  8
#define HD_   128
#define CK_   4
#define DQK_  1024
#define DV_   2048
#define NTOK_ (B_ * S_)          // 8192 tokens

__device__ __forceinline__ __bf16 f2bf(float f) {
  unsigned u = __float_as_uint(f);
  unsigned r = (u + 0x7FFFu + ((u >> 16) & 1u)) >> 16;
  unsigned short s = (unsigned short)r;
  return __builtin_bit_cast(__bf16, s);
}

// ---- CDNA5 async global->LDS copy (ASYNCcnt path), portable inline asm ----
__device__ __forceinline__ void async_copy_b128(unsigned lds_off, const void* g) {
  asm volatile("global_load_async_to_lds_b128 %0, %1, off"
               :: "v"(lds_off), "v"(g) : "memory");
}
__device__ __forceinline__ void wait_async0() {
  asm volatile("s_wait_asynccnt 0x0" ::: "memory");
}
__device__ __forceinline__ unsigned lds_addr(const void* p) {
  // shared-space flat address: LDS offset lives in addr[31:0]
  return (unsigned)(unsigned long long)p;
}

// ---------------- f32 -> bf16 convert ----------------
__global__ __launch_bounds__(256) void cvt_bf16(const float* __restrict__ in,
                                                __bf16* __restrict__ out, long n) {
  long i = (long)blockIdx.x * 256 + threadIdx.x;
  if (i < n) out[i] = f2bf(in[i]);
}

// ---------------- bf16 WMMA GEMM: C[M,N] = A[M,K] * W[N,K]^T ----------------
// Double-buffered LDS, async global->LDS staging overlapped with WMMA compute.
#define BM 128
#define BN 128
#define BK 32
#define LDSK (BK + 8)

union Frag { v16bf v; uint4 q[2]; };

template <int EPI>  // 0: none, 1: sigmoid
__global__ __launch_bounds__(256) void gemm_bf16(const __bf16* __restrict__ A,
                                                 const __bf16* __restrict__ W,
                                                 float* __restrict__ C,
                                                 int M, int N, int K) {
  __shared__ __bf16 As[2][BM][LDSK];
  __shared__ __bf16 Ws[2][BN][LDSK];
  const int tid   = threadIdx.x;
  const int lane  = tid & 31;
  const int wave  = tid >> 5;
  const int wm    = wave & 3;     // 4 wave-rows  * 32 M
  const int wn    = wave >> 2;    // 2 wave-cols  * 64 N
  const int laneM = lane & 15;
  const int kh    = lane >> 4;
  const int m0    = blockIdx.y * BM;
  const int n0    = blockIdx.x * BN;
  const int lrow  = tid >> 1;             // 0..127
  const int lhalf = (tid & 1) * 16;       // 0 or 16 (bf16 elems)

  v8f zero = {};
  v8f acc[2][4];
#pragma unroll
  for (int i = 0; i < 2; ++i)
#pragma unroll
    for (int j = 0; j < 4; ++j) acc[i][j] = zero;

  const __bf16* gA = A + (size_t)(m0 + lrow) * K + lhalf;
  const __bf16* gW = W + (size_t)(n0 + lrow) * K + lhalf;

  const unsigned bufBytes = (unsigned)(BM * LDSK * sizeof(__bf16));  // 10240
  const unsigned ldsA = lds_addr(&As[0][lrow][lhalf]);
  const unsigned ldsW = lds_addr(&Ws[0][lrow][lhalf]);

  // prologue: stage k0 = 0 into buffer 0
  {
    async_copy_b128(ldsA,      gA);
    async_copy_b128(ldsA + 16, gA + 8);
    async_copy_b128(ldsW,      gW);
    async_copy_b128(ldsW + 16, gW + 8);
  }

  int buf = 0;
  for (int k0 = 0; k0 < K; k0 += BK) {
    wait_async0();        // my stage of current buffer is done
    __syncthreads();      // everyone's stage done; prev buffer fully consumed

    if (k0 + BK < K) {    // stage next tile into other buffer (overlaps compute)
      const unsigned bo = (unsigned)(buf ^ 1) * bufBytes;
      const __bf16* pa = gA + (k0 + BK);
      const __bf16* pw = gW + (k0 + BK);
      async_copy_b128(ldsA + bo,      pa);
      async_copy_b128(ldsA + bo + 16, pa + 8);
      async_copy_b128(ldsW + bo,      pw);
      async_copy_b128(ldsW + bo + 16, pw + 8);
    }

    Frag a[2];
#pragma unroll
    for (int fm = 0; fm < 2; ++fm) {
      const __bf16* p = &As[buf][wm * 32 + fm * 16 + laneM][kh * 8];
      a[fm].q[0] = *(const uint4*)p;          // K = kh*8 .. +8
      a[fm].q[1] = *(const uint4*)(p + 16);   // K = 16+kh*8 .. +8
    }
#pragma unroll
    for (int fn = 0; fn < 4; ++fn) {
      Frag b;
      const __bf16* p = &Ws[buf][wn * 64 + fn * 16 + laneM][kh * 8];
      b.q[0] = *(const uint4*)p;
      b.q[1] = *(const uint4*)(p + 16);
#pragma unroll
      for (int fm = 0; fm < 2; ++fm) {
        acc[fm][fn] = __builtin_amdgcn_wmma_f32_16x16x32_bf16(
            false, a[fm].v, false, b.v, (short)0, acc[fm][fn], false, false);
      }
    }
    buf ^= 1;
  }

#pragma unroll
  for (int fm = 0; fm < 2; ++fm) {
#pragma unroll
    for (int fn = 0; fn < 4; ++fn) {
      const int n  = n0 + wn * 64 + fn * 16 + laneM;
      const int mb = m0 + wm * 32 + fm * 16 + kh * 8;
#pragma unroll
      for (int r = 0; r < 8; ++r) {
        float val = acc[fm][fn][r];
        if (EPI == 1) val = 1.f / (1.f + __expf(-val));
        C[(size_t)(mb + r) * N + n] = val;
      }
    }
  }
}

// ---------------- alpha/beta: sigmoid(x @ w^T + b) ----------------
__global__ __launch_bounds__(256) void alphabeta(const float* __restrict__ x,
                                                 const float* __restrict__ aw,
                                                 const float* __restrict__ ab,
                                                 const float* __restrict__ bw,
                                                 const float* __restrict__ bb,
                                                 float* __restrict__ alpha,
                                                 float* __restrict__ beta) {
  const int wave = threadIdx.x >> 5;
  const int lane = threadIdx.x & 31;
  long task = (long)blockIdx.x * 8 + wave;     // NTOK_*H_ tasks
  long tok  = task >> 4;
  int  h    = (int)(task & 15);
  const float* xp = x + tok * D_;
  const float* ap = aw + (size_t)h * D_;
  const float* bp = bw + (size_t)h * D_;
  float sa = 0.f, sb = 0.f;
  for (int i = lane; i < D_; i += 32) {
    float xv = xp[i];
    sa += xv * ap[i];
    sb += xv * bp[i];
  }
#pragma unroll
  for (int off = 16; off > 0; off >>= 1) {
    sa += __shfl_down(sa, off, 32);
    sb += __shfl_down(sb, off, 32);
  }
  if (lane == 0) {
    alpha[tok * H_ + h] = 1.f / (1.f + __expf(-(sa + ab[h])));
    beta[tok * H_ + h]  = (1.f / (1.f + __expf(-(sb + bb[h])))) * 0.08838834764831845f; // 1/sqrt(128)
  }
}

// ---------------- fused causal depthwise conv (k=4) + head RMSNorm ----------------
__global__ __launch_bounds__(128) void convnorm(const float* __restrict__ Qr,
                                                const float* __restrict__ cw,
                                                const float* __restrict__ nw,
                                                float* __restrict__ out) {
  __shared__ float red[128];
  const int  d   = threadIdx.x;
  const int  kvh = blockIdx.x % KVH_;
  const long bt  = blockIdx.x / KVH_;          // global token (b*S + t)
  const int  t   = (int)(bt % S_);
  const int  c   = kvh * HD_ + d;
  float acc = 0.f;
#pragma unroll
  for (int j = 0; j < CK_; ++j) {
    int tt = t - (CK_ - 1) + j;
    if (tt >= 0) acc += cw[c * CK_ + j] * Qr[(bt - (CK_ - 1) + j) * DQK_ + c];
  }
  red[d] = acc * acc;
  __syncthreads();
  for (int s = 64; s > 0; s >>= 1) {
    if (d < s) red[d] += red[d + s];
    __syncthreads();
  }
  float mean = red[0] * (1.f / HD_);
  out[bt * DQK_ + c] = nw[d] * acc * rsqrtf(mean + 1e-6f);
}

// ---------------- delta-rule scan: one block per (b,h), lane v owns S[:,v] ----------------
// q/k/v chunk staging uses async global->LDS b128 copies.
#define CHT 32
__global__ __launch_bounds__(128) void scan_kernel(const float* __restrict__ Qc,
                                                   const float* __restrict__ Kc,
                                                   const float* __restrict__ Vv,
                                                   const float* __restrict__ al,
                                                   const float* __restrict__ be,
                                                   float* __restrict__ O) {
  __shared__ float qs[CHT][HD_], ks[CHT][HD_], vs[CHT][HD_], as_[CHT], bs[CHT];
  const int bh  = blockIdx.x;
  const int b   = bh >> 4;
  const int h   = bh & 15;
  const int kvh = h >> 1;                      // rep = H/KVH = 2
  const int v   = threadIdx.x;
  const long tokBase = (long)b * S_;

  const unsigned qsB = lds_addr(&qs[0][0]);
  const unsigned ksB = lds_addr(&ks[0][0]);
  const unsigned vsB = lds_addr(&vs[0][0]);

  float st[HD_];
#pragma unroll
  for (int i = 0; i < HD_; ++i) st[i] = 0.f;

#pragma unroll 1
  for (int t0 = 0; t0 < S_; t0 += CHT) {
    __syncthreads();   // previous chunk fully consumed before overwrite
    // 32 rows x 128 floats per matrix = 1024 b128 slots; 8 per thread per matrix
#pragma unroll
    for (int i = 0; i < 8; ++i) {
      int f    = v + i * 128;        // 0..1023
      int row  = f >> 5;
      int slot = f & 31;             // 4-float chunk
      long tok = tokBase + t0 + row;
      unsigned lo = (unsigned)((row * HD_ + slot * 4) * 4);
      async_copy_b128(qsB + lo, &Qc[tok * DQK_ + kvh * HD_ + slot * 4]);
      async_copy_b128(ksB + lo, &Kc[tok * DQK_ + kvh * HD_ + slot * 4]);
      async_copy_b128(vsB + lo, &Vv[tok * DV_  + h   * HD_ + slot * 4]);
    }
    if (v < CHT) {
      long tok = tokBase + t0 + v;
      as_[v] = al[tok * H_ + h];
      bs[v]  = be[tok * H_ + h];
    }
    wait_async0();
    __syncthreads();

#pragma unroll 1
    for (int tt = 0; tt < CHT; ++tt) {
      float pred = 0.f;
#pragma unroll
      for (int d = 0; d < HD_; ++d) pred += ks[tt][d] * st[d];   // LDS broadcast
      float bd = bs[tt] * (vs[tt][v] - pred);
      float at = as_[tt];
      float o  = 0.f;
#pragma unroll
      for (int d = 0; d < HD_; ++d) {
        st[d] = at * st[d] + bd * ks[tt][d];
        o += qs[tt][d] * st[d];
      }
      O[(tokBase + t0 + tt) * DV_ + h * HD_ + v] = o;
    }
  }
}

// ---------------- gating: out_bf16 = sigmoid(g) * O ----------------
__global__ __launch_bounds__(256) void gate_mul(const float* __restrict__ g,
                                                const float* __restrict__ O,
                                                __bf16* __restrict__ out, long n) {
  long i = (long)blockIdx.x * 256 + threadIdx.x;
  if (i < n) out[i] = f2bf(g[i] * O[i]);
}

// ---------------- launch ----------------
extern "C" void kernel_launch(void* const* d_in, const int* in_sizes, int n_in,
                              void* d_out, int out_size, void* d_ws, size_t ws_size,
                              hipStream_t stream) {
  (void)in_sizes; (void)n_in; (void)out_size; (void)ws_size;
  const float* x   = (const float*)d_in[0];
  const float* wq  = (const float*)d_in[1];
  const float* wk  = (const float*)d_in[2];
  const float* wv  = (const float*)d_in[3];
  const float* wo  = (const float*)d_in[4];
  const float* qcw = (const float*)d_in[5];
  const float* kcw = (const float*)d_in[6];
  const float* aw  = (const float*)d_in[7];
  const float* ab  = (const float*)d_in[8];
  const float* bw  = (const float*)d_in[9];
  const float* bb  = (const float*)d_in[10];
  const float* gw  = (const float*)d_in[11];
  const float* nqw = (const float*)d_in[12];
  const float* nkw = (const float*)d_in[13];

  char* base = (char*)d_ws;
  size_t off = 0;
  auto alloc = [&](size_t bytes) {
    char* p = base + off;
    off = (off + bytes + 255) & ~(size_t)255;
    return p;
  };

  __bf16* xb  = (__bf16*)alloc((size_t)NTOK_ * D_ * 2);
  __bf16* wqb = (__bf16*)alloc((size_t)DQK_ * D_ * 2);
  __bf16* wkb = (__bf16*)alloc((size_t)DQK_ * D_ * 2);
  __bf16* wvb = (__bf16*)alloc((size_t)DV_ * D_ * 2);
  __bf16* gwb = (__bf16*)alloc((size_t)DV_ * D_ * 2);
  __bf16* wob = (__bf16*)alloc((size_t)D_ * DV_ * 2);
  float*  Qr  = (float*)alloc((size_t)NTOK_ * DQK_ * 4);
  float*  Kr  = (float*)alloc((size_t)NTOK_ * DQK_ * 4);
  float*  Qc  = (float*)alloc((size_t)NTOK_ * DQK_ * 4);
  float*  Kc  = (float*)alloc((size_t)NTOK_ * DQK_ * 4);
  float*  Vv  = (float*)alloc((size_t)NTOK_ * DV_ * 4);
  float*  Gs  = (float*)alloc((size_t)NTOK_ * DV_ * 4);
  float*  al  = (float*)alloc((size_t)NTOK_ * H_ * 4);
  float*  be  = (float*)alloc((size_t)NTOK_ * H_ * 4);
  float*  Ob  = (float*)alloc((size_t)NTOK_ * DV_ * 4);
  __bf16* Og  = (__bf16*)alloc((size_t)NTOK_ * DV_ * 2);

  // 1) bf16 conversions
  long nx = (long)NTOK_ * D_;
  cvt_bf16<<<(nx + 255) / 256, 256, 0, stream>>>(x, xb, nx);
  long nw1 = (long)DQK_ * D_, nw2 = (long)DV_ * D_;
  cvt_bf16<<<(nw1 + 255) / 256, 256, 0, stream>>>(wq, wqb, nw1);
  cvt_bf16<<<(nw1 + 255) / 256, 256, 0, stream>>>(wk, wkb, nw1);
  cvt_bf16<<<(nw2 + 255) / 256, 256, 0, stream>>>(wv, wvb, nw2);
  cvt_bf16<<<(nw2 + 255) / 256, 256, 0, stream>>>(gw, gwb, nw2);
  cvt_bf16<<<(nw2 + 255) / 256, 256, 0, stream>>>(wo, wob, nw2);

  // 2) projection GEMMs (WMMA bf16, async double-buffered staging)
  dim3 gq(DQK_ / BN, NTOK_ / BM);
  gemm_bf16<0><<<gq, 256, 0, stream>>>(xb, wqb, Qr, NTOK_, DQK_, D_);
  gemm_bf16<0><<<gq, 256, 0, stream>>>(xb, wkb, Kr, NTOK_, DQK_, D_);
  dim3 gv(DV_ / BN, NTOK_ / BM);
  gemm_bf16<0><<<gv, 256, 0, stream>>>(xb, wvb, Vv, NTOK_, DV_, D_);
  gemm_bf16<1><<<gv, 256, 0, stream>>>(xb, gwb, Gs, NTOK_, DV_, D_);  // sigmoid fused

  // 3) gates
  alphabeta<<<(NTOK_ * H_) / 8, 256, 0, stream>>>(x, aw, ab, bw, bb, al, be);

  // 4) conv + rmsnorm
  convnorm<<<NTOK_ * KVH_, 128, 0, stream>>>(Qr, qcw, nqw, Qc);
  convnorm<<<NTOK_ * KVH_, 128, 0, stream>>>(Kr, kcw, nkw, Kc);

  // 5) sequential delta-rule scan
  scan_kernel<<<B_ * H_, 128, 0, stream>>>(Qc, Kc, Vv, al, be, Ob);

  // 6) gating -> bf16
  long nv = (long)NTOK_ * DV_;
  gate_mul<<<(nv + 255) / 256, 256, 0, stream>>>(Gs, Ob, Og, nv);

  // 7) output GEMM -> d_out (f32)
  dim3 go(D_ / BN, NTOK_ / BM);
  gemm_bf16<0><<<go, 256, 0, stream>>>(Og, wob, (float*)d_out, NTOK_, D_, DV_);
}
